// MoeLayer_35304631173960
// MI455X (gfx1250) — matrix operand order repaired and checked by reading, not compiled, not run on hardware
//
#include <hip/hip_runtime.h>
#include <hip/hip_bf16.h>

// ---------------------------------------------------------------------------
// MoE layer for MI455X (gfx1250): top-2 dispatch + bf16 WMMA expert MLP.
// Weights pre-transposed in global memory so all LDS staging is contiguous
// 16B segments -> global_load_async_to_lds_b128 (ASYNCcnt).
// ---------------------------------------------------------------------------

#define MOE_N 16384
#define MOE_D 256
#define MOE_E 8
#define MOE_H 512
#define MOE_TM 64          // tokens per FFN tile
#define XS_STRIDE 264      // 132 dwords/row -> conflict-friendly, 16B aligned
#define W1_STRIDE 264      // [hcol][k] stage of W1 chunk
#define W2_STRIDE 72       // [dcol][k] stage of W2 chunk (36 dwords)
#define HT_STRIDE 72       // [m][k] staged GELU output

typedef __bf16 bf16_t;
typedef __attribute__((ext_vector_type(8)))  __bf16 v8bf;
typedef __attribute__((ext_vector_type(16))) __bf16 v16bf;
typedef __attribute__((ext_vector_type(8)))  float  v8f;

#if defined(__AMDGCN__) &&                                                     \
    __has_builtin(__builtin_amdgcn_global_load_async_to_lds_b128) &&           \
    __has_builtin(__builtin_amdgcn_s_wait_asynccnt)
#define MOE_ASYNC_LDS 1
#endif

#ifdef MOE_ASYNC_LDS
// Builtin signature (verified via hipcc diagnostic): (int4 AS1*, int4 AS3*,
// imm offset, imm cpol).
typedef int v4i_gcc __attribute__((vector_size(16)));
typedef __attribute__((address_space(1))) v4i_gcc gv4i;
typedef __attribute__((address_space(3))) v4i_gcc lv4i;
#endif

// 16B global -> LDS copy (async engine; per-lane addresses).
__device__ __forceinline__ void cp16_g2l(const bf16_t* g, bf16_t* l) {
#ifdef MOE_ASYNC_LDS
  __builtin_amdgcn_global_load_async_to_lds_b128(
      (gv4i*)(uintptr_t)g, (lv4i*)(uint32_t)(uintptr_t)l, 0, 0);
#else
  *(v8bf*)l = *(const v8bf*)g;
#endif
}

__device__ __forceinline__ void cp_wait() {
#ifdef MOE_ASYNC_LDS
  __builtin_amdgcn_s_wait_asynccnt(0);
#endif
}

// Scheduling fence: keep LDS fragment loads issued before the WMMA group so
// the backend emits incremental s_wait_dscnt instead of per-WMMA dscnt==0.
__device__ __forceinline__ void sched_fence() {
#if defined(__AMDGCN__) && __has_builtin(__builtin_amdgcn_sched_barrier)
  __builtin_amdgcn_sched_barrier(0);
#endif
}

__device__ __forceinline__ v16bf cat8(v8bf lo, v8bf hi) {
  return __builtin_shufflevector(lo, hi, 0,1,2,3,4,5,6,7,8,9,10,11,12,13,14,15);
}

// A-fragment (16x32 bf16) from row-major LDS [m][k]:
// lanes 0-15: M=lane, elems 0-7 = K+0..7,  elems 8-15 = K+16..23
// lanes16-31: M=lane-16, elems 0-7 = K+8..15, elems 8-15 = K+24..31
__device__ __forceinline__ v16bf load_a(const bf16_t* s, int stride, int mbase,
                                        int kbase, int lane) {
  int row  = mbase + (lane & 15);
  int half = lane >> 4;
  const bf16_t* p = s + row * stride + kbase + half * 8;
  v8bf lo = *(const v8bf*)(p);
  v8bf hi = *(const v8bf*)(p + 16);
  return cat8(lo, hi);
}

// B-fragment (32x16 bf16) from column-major LDS stage [n][k]:
// lanes 0-15: N=lane, elems 0-7 = K+0..7, elems 8-15 = K+16..23 ; lanes 16-31: N=lane-16, elems 0-7 = K+8..15, elems 8-15 = K+24..31
__device__ __forceinline__ v16bf load_b(const bf16_t* s, int kstride, int nbase,
                                        int kbase, int lane) {
  int col  = nbase + (lane & 15);
  int koff = kbase + ((lane >> 4) << 3);
  const bf16_t* p = s + col * kstride + koff;
  v8bf lo = *(const v8bf*)(p);
  v8bf hi = *(const v8bf*)(p + 16);
  return cat8(lo, hi);
}

__device__ __forceinline__ v8f wmma_bf16(v16bf a, v16bf b, v8f c) {
  return __builtin_amdgcn_wmma_f32_16x16x32_bf16(false, a, false, b, (short)0,
                                                 c, false, false);
}

__device__ __forceinline__ float gelu_exact(float x) {
  return 0.5f * x * (1.0f + erff(x * 0.70710678118654752f));
}

// --------------------------- utility kernels -------------------------------

__global__ void k_zero_f32(float* p, int n) {
  for (int i = blockIdx.x * blockDim.x + threadIdx.x; i < n;
       i += gridDim.x * blockDim.x)
    p[i] = 0.0f;
}

__global__ void k_zero_i32(int* p, int n) {
  int i = blockIdx.x * blockDim.x + threadIdx.x;
  if (i < n) p[i] = 0;
}

__global__ void k_cast_bf16(const float* __restrict__ s, bf16_t* __restrict__ d,
                            int n) {
  for (int i = blockIdx.x * blockDim.x + threadIdx.x; i < n;
       i += gridDim.x * blockDim.x)
    d[i] = (bf16_t)s[i];
}

// Transpose+cast: per matrix m: src [R][C] f32 -> dst [C][R] bf16.
// grid (C/32, R/32, nmat), block (32, 8). Coalesced both sides via LDS tile.
__global__ __launch_bounds__(256) void k_transpose_cast(
    const float* __restrict__ s, bf16_t* __restrict__ d, int R, int C) {
  __shared__ float tile[32][33];
  const float* sm = s + (size_t)blockIdx.z * R * C;
  bf16_t*      dm = d + (size_t)blockIdx.z * R * C;
  int c0 = blockIdx.x * 32;
  int r0 = blockIdx.y * 32;
  int tx = threadIdx.x, ty = threadIdx.y;
#pragma unroll
  for (int i = 0; i < 32; i += 8)
    tile[ty + i][tx] = sm[(size_t)(r0 + ty + i) * C + c0 + tx];
  __syncthreads();
#pragma unroll
  for (int i = 0; i < 32; i += 8)
    dm[(size_t)(c0 + ty + i) * R + r0 + tx] = (bf16_t)tile[tx][ty + i];
}

// ------------------------------- gating ------------------------------------
__global__ __launch_bounds__(256) void k_gate(const float* __restrict__ x,
                                              const float* __restrict__ Wg,
                                              int* __restrict__ counts,
                                              int* __restrict__ tok_idx,
                                              float* __restrict__ tok_w) {
  __shared__ float WgS[MOE_D * MOE_E];  // [d][e], 8 KB
  int tid = threadIdx.x;
  for (int i = tid; i < MOE_D * MOE_E; i += 256) WgS[i] = Wg[i];
  __syncthreads();

  int n = blockIdx.x * 256 + tid;
  float acc[MOE_E];
#pragma unroll
  for (int e = 0; e < MOE_E; ++e) acc[e] = 0.0f;

  const float4* xr = (const float4*)(x + (size_t)n * MOE_D);
#pragma unroll 4
  for (int d4 = 0; d4 < MOE_D / 4; ++d4) {
    float4 v = xr[d4];
    int d = d4 * 4;
#pragma unroll
    for (int e = 0; e < MOE_E; ++e) {
      acc[e] += v.x * WgS[(d + 0) * MOE_E + e];
      acc[e] += v.y * WgS[(d + 1) * MOE_E + e];
      acc[e] += v.z * WgS[(d + 2) * MOE_E + e];
      acc[e] += v.w * WgS[(d + 3) * MOE_E + e];
    }
  }

  int i0 = 0;
#pragma unroll
  for (int e = 1; e < MOE_E; ++e)
    if (acc[e] > acc[i0]) i0 = e;
  int i1 = (i0 == 0) ? 1 : 0;
#pragma unroll
  for (int e = 0; e < MOE_E; ++e)
    if (e != i0 && acc[e] > acc[i1]) i1 = e;

  float m  = acc[i0];
  float e0 = expf(acc[i0] - m);
  float e1 = expf(acc[i1] - m);
  float inv = 1.0f / (e0 + e1);

  int p0 = atomicAdd(&counts[i0], 1);
  tok_idx[i0 * MOE_N + p0] = n;
  tok_w[i0 * MOE_N + p0]   = e0 * inv;
  int p1 = atomicAdd(&counts[i1], 1);
  tok_idx[i1 * MOE_N + p1] = n;
  tok_w[i1 * MOE_N + p1]   = e1 * inv;
}

// --------------------------- expert FFN (WMMA) -----------------------------
// grid.x = E * (N/TM). Block: 256 threads = 8 wave32.
__global__ __launch_bounds__(256) void k_ffn(const bf16_t* __restrict__ X16,
                                             const bf16_t* __restrict__ W1T,
                                             const bf16_t* __restrict__ W2T,
                                             const float* __restrict__ b1,
                                             const float* __restrict__ b2,
                                             const int* __restrict__ counts,
                                             const int* __restrict__ tok_idx,
                                             const float* __restrict__ tok_w,
                                             float* __restrict__ out) {
  const int tilesPerE = MOE_N / MOE_TM;  // 256
  int e = blockIdx.x / tilesPerE;
  int t = blockIdx.x % tilesPerE;
  int cnt  = counts[e];
  int row0 = t * MOE_TM;
  if (row0 >= cnt) return;  // uniform per block
  int mrows = min(MOE_TM, cnt - row0);

  extern __shared__ char smem[];
  bf16_t* Xs   = (bf16_t*)smem;              // [64][XS_STRIDE] row-major [m][k]
  bf16_t* W1sT = Xs + 64 * XS_STRIDE;        // [64][W1_STRIDE] [h][k]
  bf16_t* W2sT = W1sT + 64 * W1_STRIDE;      // [256][W2_STRIDE] [d][k]
  bf16_t* Htmp = W2sT + 256 * W2_STRIDE;     // [64][HT_STRIDE] [m][k]

  int tid  = threadIdx.x;
  int lane = tid & 31;
  int wave = tid >> 5;

  const bf16_t* W1Te = W1T + (size_t)e * MOE_H * MOE_D;  // [H][D]: [h][k]
  const bf16_t* W2Te = W2T + (size_t)e * MOE_D * MOE_H;  // [D][H]: [d][k]

  // ---- stage X tile (gathered token rows), 16B async segments ----
#pragma unroll
  for (int it = 0; it < 8; ++it) {
    int seg = tid + it * 256;        // 64 rows x 32 segs
    int r = seg >> 5, s = seg & 31;
    int rc  = min(r, mrows - 1);     // clamp; invalid rows masked in epilogue
    int tok = tok_idx[e * MOE_N + row0 + rc];
    cp16_g2l(X16 + (size_t)tok * MOE_D + s * 8, Xs + r * XS_STRIDE + s * 8);
  }

  int mstrip = wave >> 1;   // stage-2 rows: mstrip*16..+16
  int nhalf  = wave & 1;    // stage-2 cols: nhalf*128..+128
  v8f acc2[8];
#pragma unroll
  for (int i = 0; i < 8; ++i) acc2[i] = (v8f){};

  cp_wait();
  __syncthreads();

#pragma unroll 1
  for (int hc = 0; hc < 8; ++hc) {
    int hbase = hc * 64;

    // ---- stage W1 chunk: rows h in [hbase,+64), 256 k each (contig 16B) ----
#pragma unroll
    for (int it = 0; it < 8; ++it) {
      int seg = tid + it * 256;      // 64 rows x 32 segs
      int r = seg >> 5, s = seg & 31;
      cp16_g2l(W1Te + (size_t)(hbase + r) * MOE_D + s * 8,
               W1sT + r * W1_STRIDE + s * 8);
    }
    // ---- stage W2 chunk: 256 d-rows x 64 k (contig 16B) ----
#pragma unroll
    for (int it = 0; it < 8; ++it) {
      int seg = tid + it * 256;      // 256 rows x 8 segs
      int r = seg >> 3, s = seg & 7;
      cp16_g2l(W2Te + (size_t)r * MOE_H + hbase + s * 8,
               W2sT + r * W2_STRIDE + s * 8);
    }
    if (hc + 1 < 8)
      __builtin_prefetch(W1Te + (size_t)(hbase + 64) * MOE_D, 0, 0);
    cp_wait();
    __syncthreads();

    // ---- stage 1: Htile[64x64] = Xs[64x256] @ W1chunk (2-deep pipeline) ----
    {
      int ms  = (wave >> 1) * 16;
      int n0w = (wave & 1) * 32;
      v8f c0 = (v8f){}, c1 = (v8f){};
      v16bf a   = load_a(Xs, XS_STRIDE, ms, 0, lane);
      v16bf b0  = load_b(W1sT, W1_STRIDE, n0w, 0, lane);
      v16bf b1v = load_b(W1sT, W1_STRIDE, n0w + 16, 0, lane);
#pragma unroll
      for (int kc = 0; kc < 8; ++kc) {
        v16bf an  = (kc < 7) ? load_a(Xs, XS_STRIDE, ms, (kc + 1) * 32, lane) : a;
        v16bf b0n = (kc < 7) ? load_b(W1sT, W1_STRIDE, n0w, (kc + 1) * 32, lane) : b0;
        v16bf b1n = (kc < 7) ? load_b(W1sT, W1_STRIDE, n0w + 16, (kc + 1) * 32, lane) : b1v;
        sched_fence();  // keep next-iter loads above this iter's WMMAs
        c0 = wmma_bf16(a, b0, c0);
        c1 = wmma_bf16(a, b1v, c1);
        sched_fence();
        a = an; b0 = b0n; b1v = b1n;
      }
      // bias + exact GELU -> bf16 -> Htmp
      int half = lane >> 4;
      int nl   = lane & 15;
      float bA = b1[e * MOE_H + hbase + n0w + nl];
      float bB = b1[e * MOE_H + hbase + n0w + 16 + nl];
#pragma unroll
      for (int v = 0; v < 8; ++v) {
        int rr = ms + v + half * 8;
        float x0 = gelu_exact(c0[v] + bA);
        float x1 = gelu_exact(c1[v] + bB);
        Htmp[rr * HT_STRIDE + n0w + nl]      = (bf16_t)x0;
        Htmp[rr * HT_STRIDE + n0w + 16 + nl] = (bf16_t)x1;
      }
    }
    __syncthreads();

    // ---- stage 2: hoist A + all 8 B tiles, then 8 back-to-back WMMAs ----
#pragma unroll
    for (int kc = 0; kc < 2; ++kc) {
      v16bf a = load_a(Htmp, HT_STRIDE, mstrip * 16, kc * 32, lane);
      v16bf bt[8];
#pragma unroll
      for (int nt = 0; nt < 8; ++nt)
        bt[nt] = load_b(W2sT, W2_STRIDE, nhalf * 128 + nt * 16, kc * 32, lane);
      sched_fence();  // loads stay above; WMMAs below get incremental waits
#pragma unroll
      for (int nt = 0; nt < 8; ++nt)
        acc2[nt] = wmma_bf16(a, bt[nt], acc2[nt]);
      sched_fence();
    }
    __syncthreads();
  }

  // ---- epilogue: +b2, scale by routing weight, atomic add into output ----
  int half = lane >> 4;
  int nl   = lane & 15;
  float bias2[8];
#pragma unroll
  for (int nt = 0; nt < 8; ++nt)
    bias2[nt] = b2[e * MOE_D + nhalf * 128 + nt * 16 + nl];
#pragma unroll
  for (int v = 0; v < 8; ++v) {
    int rl = mstrip * 16 + v + half * 8;
    if (row0 + rl < cnt) {
      int   tok = tok_idx[e * MOE_N + row0 + rl];
      float wgt = tok_w[e * MOE_N + row0 + rl];
      float* orow = out + (size_t)tok * MOE_D;
#pragma unroll
      for (int nt = 0; nt < 8; ++nt) {
        int col = nhalf * 128 + nt * 16 + nl;
        atomicAdd(orow + col, wgt * (acc2[nt][v] + bias2[nt]));
      }
    }
  }
}

// ------------------------------- launcher ----------------------------------

extern "C" void kernel_launch(void* const* d_in, const int* in_sizes, int n_in,
                              void* d_out, int out_size, void* d_ws,
                              size_t ws_size, hipStream_t stream) {
  const float* inputs = (const float*)d_in[0];
  const float* Wg     = (const float*)d_in[1];
  const float* w1     = (const float*)d_in[2];
  const float* b1     = (const float*)d_in[3];
  const float* w2     = (const float*)d_in[4];
  const float* b2     = (const float*)d_in[5];
  float* out = (float*)d_out;

  char* ws = (char*)d_ws;
  size_t offX   = 0;
  size_t offW1  = offX  + (size_t)MOE_N * MOE_D * sizeof(bf16_t);
  size_t offW2  = offW1 + (size_t)MOE_E * MOE_H * MOE_D * sizeof(bf16_t);
  size_t offCnt = offW2 + (size_t)MOE_E * MOE_D * MOE_H * sizeof(bf16_t);
  size_t offIdx = offCnt + 256;
  size_t offWt  = offIdx + (size_t)MOE_E * MOE_N * sizeof(int);

  bf16_t* X16  = (bf16_t*)(ws + offX);
  bf16_t* W1T  = (bf16_t*)(ws + offW1);  // [E][H][D]
  bf16_t* W2T  = (bf16_t*)(ws + offW2);  // [E][D][H]
  int*    cnts = (int*)(ws + offCnt);
  int*    tIdx = (int*)(ws + offIdx);
  float*  tW   = (float*)(ws + offWt);

  k_zero_i32<<<1, 32, 0, stream>>>(cnts, MOE_E);
  k_zero_f32<<<1024, 256, 0, stream>>>(out, MOE_N * MOE_D);

  k_cast_bf16<<<1024, 256, 0, stream>>>(inputs, X16, MOE_N * MOE_D);
  // w1 [E][D][H] -> W1T [E][H][D]
  k_transpose_cast<<<dim3(MOE_H / 32, MOE_D / 32, MOE_E), dim3(32, 8), 0,
                     stream>>>(w1, W1T, MOE_D, MOE_H);
  // w2 [E][H][D] -> W2T [E][D][H]
  k_transpose_cast<<<dim3(MOE_D / 32, MOE_H / 32, MOE_E), dim3(32, 8), 0,
                     stream>>>(w2, W2T, MOE_H, MOE_D);

  k_gate<<<MOE_N / 256, 256, 0, stream>>>(inputs, Wg, cnts, tIdx, tW);

  size_t ldsBytes = (size_t)(64 * XS_STRIDE + 64 * W1_STRIDE +
                             256 * W2_STRIDE + 64 * HT_STRIDE) * sizeof(bf16_t);
  k_ffn<<<MOE_E * (MOE_N / MOE_TM), 256, ldsBytes, stream>>>(
      X16, W1T, W2T, b1, b2, cnts, tIdx, tW, out);
}
